// S_LSU_5970004541949
// MI455X (gfx1250) — matrix operand order repaired and checked by reading, not compiled, not run on hardware
//
#include <hip/hip_runtime.h>

typedef unsigned short u16;
typedef unsigned int u32;
typedef __bf16 bf16x16 __attribute__((ext_vector_type(16)));
typedef float f32x8 __attribute__((ext_vector_type(8)));
typedef int int4v __attribute__((vector_size(16)));  // matches async-LDS builtin param

union FragBF {
    bf16x16 bf;
    uint4 q[2];
};

#define NN 32
#define CC 64
#define TT 256
#define VV 25
#define HH 8
#define ICN 16
#define OCN 64
#define TV 6400          // T*V
#define CHT 8            // t per chunk
#define CHCOLS 200       // CHT*V
#define BN_CNT 204800.0f // N*T*V
#define EPS 1e-5f

// ---------------------------------------------------------------------------
// gfx1250 async global->LDS copy (ASYNCcnt-tracked DMA), guarded so the file
// still compiles if the builtin is absent (falls back to sync copy).
// Signature (from compiler diagnostic): (int4v AS(1)*, int4v AS(3)*, Ii, Ii)
// ---------------------------------------------------------------------------
#if defined(__has_builtin)
#if __has_builtin(__builtin_amdgcn_global_load_async_to_lds_b128)
#define HAVE_ASYNC_LDS 1
#endif
#endif

__device__ __forceinline__ void async_cp16(void* lds_dst, const void* gsrc) {
#if defined(HAVE_ASYNC_LDS)
    __builtin_amdgcn_global_load_async_to_lds_b128(
        (__attribute__((address_space(1))) int4v*)gsrc,
        (__attribute__((address_space(3))) int4v*)lds_dst, 0, 0);
#else
    *(uint4*)lds_dst = *(const uint4*)gsrc;
#endif
}

__device__ __forceinline__ void wait_async() {
#if defined(HAVE_ASYNC_LDS)
#if __has_builtin(__builtin_amdgcn_s_wait_asynccnt)
    __builtin_amdgcn_s_wait_asynccnt(0);
#else
    asm volatile("s_wait_asynccnt 0" ::: "memory");
#endif
#endif
}

__device__ __forceinline__ u16 f2bf(float f) {
    u32 u = __float_as_uint(f);
    u32 r = (u + 0x7fffu + ((u >> 16) & 1u)) >> 16;
    return (u16)r;
}
__device__ __forceinline__ u32 pack2(float lo, float hi) {
    return (u32)f2bf(lo) | ((u32)f2bf(hi) << 16);
}

// ---------------------------------------------------------------------------
// K0: x f32 -> bf16, two layouts:
//   xb_cm [n][t*25+v][64]   (column-major cols for GEMM1 B-fragments)
//   xb_pad[n][c][t][32]     (v padded to 32 with zeros, for agg A-fragments)
// ---------------------------------------------------------------------------
__global__ __launch_bounds__(64) void k_convert_x(const float* __restrict__ x,
                                                  u16* __restrict__ xb_cm,
                                                  u16* __restrict__ xb_pad) {
    int nt = blockIdx.x;           // 0..N*T-1
    int n = nt >> 8, t = nt & 255;
    int c = threadIdx.x;
    const float* px = x + (((size_t)n * CC + c) * TT + t) * VV;
    u16* pp = xb_pad + (((size_t)n * CC + c) * TT + t) * 32;
    u16* pc = xb_cm + ((size_t)n * TV + t * VV) * 64 + c;
#pragma unroll
    for (int v = 0; v < VV; ++v) {
        u16 b = f2bf(px[v]);
        pp[v] = b;
        pc[(size_t)v * 64] = b;
    }
#pragma unroll
    for (int v = VV; v < 32; ++v) pp[v] = 0;
}

// ---------------------------------------------------------------------------
// K0w: weights -> bf16. P[h][32][64] = stack(Wa,Wb); Wdb[h][64][64];
//      biasAB[h][32]; bdsum[64]
// ---------------------------------------------------------------------------
__global__ __launch_bounds__(256) void k_convert_w(
    const float* __restrict__ Wa, const float* __restrict__ Wb,
    const float* __restrict__ ba, const float* __restrict__ bb,
    const float* __restrict__ Wd, const float* __restrict__ bd,
    u16* __restrict__ P, u16* __restrict__ Wdb,
    float* __restrict__ biasAB, float* __restrict__ bdsum) {
    int idx = blockIdx.x * 256 + threadIdx.x;
    if (idx < 16384) { // P: 8*32*64
        int h = idx >> 11, j = (idx >> 6) & 31, c = idx & 63;
        float v = (j < 16) ? Wa[(h * 16 + j) * 64 + c] : Wb[(h * 16 + (j - 16)) * 64 + c];
        P[idx] = f2bf(v);
    } else if (idx < 49152) { // Wdb: 8*64*64
        int k = idx - 16384;
        Wdb[k] = f2bf(Wd[k]);
    } else if (idx < 49408) { // biasAB: 8*32
        int k = idx - 49152, h = k >> 5, j = k & 31;
        biasAB[k] = (j < 16) ? ba[h * 16 + j] : bb[h * 16 + (j - 16)];
    } else if (idx < 49472) { // bdsum: 64
        int o = idx - 49408;
        float s = 0.f;
        for (int h = 0; h < HH; ++h) s += bd[h * 64 + o];
        bdsum[o] = s;
    }
}

// ---------------------------------------------------------------------------
// K1: per (n,h): scores[v][w] = (1/(IC*T)) * sum_{i,t} a[i,t,v]*b[i,t,w]
// ab chunk computed via WMMA (P 32x64 x xT-chunk 64x200), stored transposed
// in LDS (abT[col][j]) so scores fragments are contiguous LDS reads.
// x-chunks are double-buffered and staged with async global->LDS DMA that
// overlaps the WMMA work on the other buffer.
// ---------------------------------------------------------------------------
__global__ __launch_bounds__(128) void k_scores(const u16* __restrict__ xb_cm,
                                                const u16* __restrict__ P,
                                                const float* __restrict__ biasAB,
                                                float* __restrict__ scorespad) {
    int n = blockIdx.x >> 3, h = blockIdx.x & 7;
    int tid = threadIdx.x;
    int wv = tid >> 5;
    int lane = tid & 31;
    int l = lane & 15;
    int hi = lane >> 4;

    __shared__ __align__(16) u16 sP[32][64];        // 4 KB
    __shared__ __align__(16) u16 sXT[2][208][64];   // 53.2 KB  row=col(t,v), 64 c
    __shared__ __align__(16) u16 sAbT[208][32];     // 13.3 KB  row=col(t,v), 32 j

    // stage P[h] (2048 u16 = 256 uint4)
    {
        const uint4* gP = (const uint4*)(P + (size_t)h * 32 * 64);
        uint4* lp = (uint4*)&sP[0][0];
        lp[tid] = gP[tid];
        lp[tid + 128] = gP[tid + 128];
    }
    // zero xT pad rows (200..207) of both buffers
    for (int b = 0; b < 2; ++b)
        for (int i = tid; i < 8 * 64; i += 128) sXT[b][200 + (i >> 6)][i & 63] = 0;
    for (int i = tid; i < 8 * 32; i += 128) sAbT[200 + (i >> 5)][i & 31] = 0;

    // issue async copy of chunk 0 into buffer 0 (1600 uint4, contiguous)
    {
        const uint4* gx = (const uint4*)(xb_cm + (size_t)n * TV * 64);
        uint4* lx = (uint4*)&sXT[0][0][0];
        for (int i = tid; i < 1600; i += 128) async_cp16(lx + i, gx + i);
    }

    int smt = wv >> 1, snt = wv & 1;   // this wave's 16x16 scores tile
    f32x8 accS = {0.f, 0.f, 0.f, 0.f, 0.f, 0.f, 0.f, 0.f};

    for (int ch = 0; ch < TT / CHT; ++ch) {
        int buf = ch & 1;
        wait_async();       // my async transfers (into buf) complete
        __syncthreads();    // everyone's transfers done; prev scores reads of sAbT done

        // prefetch next chunk into the other buffer (overlaps WMMA below)
        if (ch + 1 < TT / CHT) {
            const uint4* gx = (const uint4*)(xb_cm + ((size_t)n * TV + (ch + 1) * CHCOLS) * 64);
            uint4* lx = (uint4*)&sXT[buf ^ 1][0][0];
            for (int i = tid; i < 1600; i += 128) async_cp16(lx + i, gx + i);
        }

        // ab = P(32x64) x xT(64x200+pad): N-tiles round-robin over waves
        for (int nt = wv; nt < 13; nt += 4) {
#pragma unroll
            for (int mt = 0; mt < 2; ++mt) {
                f32x8 c;
#pragma unroll
                for (int r = 0; r < 8; ++r) c[r] = biasAB[h * 32 + mt * 16 + hi * 8 + r];
#pragma unroll
                for (int kt = 0; kt < 2; ++kt) {
                    FragBF a, b;
                    const u16* pa = &sP[mt * 16 + l][kt * 32 + hi * 8];
                    a.q[0] = *(const uint4*)pa;
                    a.q[1] = *(const uint4*)(pa + 16);
                    const u16* pb = &sXT[buf][nt * 16 + l][kt * 32 + hi * 8];
                    b.q[0] = *(const uint4*)pb;
                    b.q[1] = *(const uint4*)(pb + 16);
                    c = __builtin_amdgcn_wmma_f32_16x16x32_bf16(false, a.bf, false, b.bf,
                                                               (short)0, c, false, false);
                }
                // store transposed: abT[col][j], 8 contiguous bf16 per lane
                uint4 q;
                q.x = pack2(c[0], c[1]);
                q.y = pack2(c[2], c[3]);
                q.z = pack2(c[4], c[5]);
                q.w = pack2(c[6], c[7]);
                *(uint4*)&sAbT[nt * 16 + l][mt * 16 + hi * 8] = q;
            }
        }
        __syncthreads();

        // scores += aT x b over K=(i,tl): 4 K-tiles of 32 per chunk
#pragma unroll
        for (int kt = 0; kt < 4; ++kt) {
            FragBF a, b;
            const u16* pa0 = &sAbT[(2 * kt) * VV + smt * 16 + l][hi * 8];
            const u16* pa1 = &sAbT[(2 * kt + 1) * VV + smt * 16 + l][hi * 8];
            a.q[0] = *(const uint4*)pa0;
            a.q[1] = *(const uint4*)pa1;
            const u16* pb0 = &sAbT[(2 * kt) * VV + snt * 16 + l][16 + hi * 8];
            const u16* pb1 = &sAbT[(2 * kt + 1) * VV + snt * 16 + l][16 + hi * 8];
            b.q[0] = *(const uint4*)pb0;
            b.q[1] = *(const uint4*)pb1;
            accS = __builtin_amdgcn_wmma_f32_16x16x32_bf16(false, a.bf, false, b.bf,
                                                           (short)0, accS, false, false);
        }
    }

    const float scale = 1.0f / (float)(ICN * TT);
    float* outp = scorespad + ((size_t)(n * HH + h)) * 1024;
#pragma unroll
    for (int r = 0; r < 8; ++r) {
        int v = smt * 16 + hi * 8 + r;
        int w = snt * 16 + l;
        outp[v * 32 + w] = accS[r] * scale;
    }
}

// ---------------------------------------------------------------------------
// K2: softmax over v (axis=-2) per column w, + DepM; store transposed/padded
// bf16: attnT[n][h][w][v] with zero pad (v>=25 or w>=25).
// ---------------------------------------------------------------------------
__global__ __launch_bounds__(32) void k_softmax(const float* __restrict__ scorespad,
                                                const float* __restrict__ DepM,
                                                u16* __restrict__ attnT) {
    int n = blockIdx.x >> 3, h = blockIdx.x & 7;
    int w = threadIdx.x;
    const float* sp = scorespad + ((size_t)(n * HH + h)) * 1024;
    u16* at = attnT + ((size_t)(n * HH + h)) * 1024;
    float col[VV];
    if (w < VV) {
        float m = -3.0e38f;
#pragma unroll
        for (int v = 0; v < VV; ++v) {
            col[v] = sp[v * 32 + w];
            m = fmaxf(m, col[v]);
        }
        float s = 0.f;
#pragma unroll
        for (int v = 0; v < VV; ++v) {
            col[v] = __expf(col[v] - m);
            s += col[v];
        }
        float inv = 1.0f / s;
#pragma unroll
        for (int v = 0; v < VV; ++v)
            col[v] = col[v] * inv + DepM[((size_t)h * VV + v) * VV + w];
    }
#pragma unroll
    for (int v = 0; v < 32; ++v) {
        float val = (w < VV && v < VV) ? col[v] : 0.0f;
        at[w * 32 + v] = f2bf(val);
    }
}

// ---------------------------------------------------------------------------
// K3: per (n, t-chunk): z[o,(t,w)] = sum_h Wd_h x (x_chunk x attn_h) + bdsum
// agg GEMM K=v(pad 32); output GEMM K=c (2 tiles). z accumulated in regs
// across all heads; written to d_out (used as z scratch before BN).
// x chunk staged via async DMA, waited before first use.
// ---------------------------------------------------------------------------
__global__ __launch_bounds__(128) void k_agg_out(const u16* __restrict__ xb_pad,
                                                 const u16* __restrict__ attnT,
                                                 const u16* __restrict__ Wdb,
                                                 const float* __restrict__ bdsum,
                                                 float* __restrict__ zout) {
    int n = blockIdx.x;
    int tc = blockIdx.y;  // 8-t chunk
    int tid = threadIdx.x;
    int wv = tid >> 5;
    int lane = tid & 31;
    int l = lane & 15;
    int hi = lane >> 4;

    __shared__ __align__(16) u16 sX[64][256];   // [c][t*32+v]  32 KB
    __shared__ __align__(16) u16 sYT[256][64];  // [col][c]     32 KB
    __shared__ __align__(16) u16 sAt[32][32];   // attnT tile    2 KB

    // async stage of x chunk: 64 rows x 32 uint4 (row base strided in global)
    for (int i = tid; i < 2048; i += 128) {
        int c = i >> 5, j = i & 31;
        async_cp16(((uint4*)&sX[c][0]) + j,
                   ((const uint4*)(xb_pad + (((size_t)n * CC + c) * TT + tc * CHT) * 32)) + j);
    }

    f32x8 acc[4][4];  // [mt][nti], nt = wv + nti*4
#pragma unroll
    for (int mt = 0; mt < 4; ++mt)
#pragma unroll
        for (int nti = 0; nti < 4; ++nti)
#pragma unroll
            for (int r = 0; r < 8; ++r) acc[mt][nti][r] = 0.f;

    for (int h = 0; h < HH; ++h) {
        __syncthreads();
        // stage attnT tile (1024 u16 = 128 uint4)
        ((uint4*)&sAt[0][0])[tid] = ((const uint4*)(attnT + ((size_t)(n * HH + h)) * 1024))[tid];
        if (h == 0) wait_async();  // x chunk resident before first agg
        __syncthreads();

        // agg: y[c, w] per t; this wave handles t = wv*2 + {0,1}
#pragma unroll
        for (int tt = 0; tt < 2; ++tt) {
            int t = wv * 2 + tt;
#pragma unroll
            for (int mt = 0; mt < 4; ++mt) {
#pragma unroll
                for (int nt2 = 0; nt2 < 2; ++nt2) {
                    FragBF a, b;
                    f32x8 d = {0.f, 0.f, 0.f, 0.f, 0.f, 0.f, 0.f, 0.f};
                    const u16* pa = &sX[mt * 16 + l][t * 32 + hi * 8];
                    a.q[0] = *(const uint4*)pa;
                    a.q[1] = *(const uint4*)(pa + 16);
                    const u16* pb = &sAt[nt2 * 16 + l][hi * 8];
                    b.q[0] = *(const uint4*)pb;
                    b.q[1] = *(const uint4*)(pb + 16);
                    d = __builtin_amdgcn_wmma_f32_16x16x32_bf16(false, a.bf, false, b.bf,
                                                                (short)0, d, false, false);
                    uint4 q;
                    q.x = pack2(d[0], d[1]);
                    q.y = pack2(d[2], d[3]);
                    q.z = pack2(d[4], d[5]);
                    q.w = pack2(d[6], d[7]);
                    *(uint4*)&sYT[t * 32 + nt2 * 16 + l][mt * 16 + hi * 8] = q;
                }
            }
        }
        __syncthreads();

        // z += Wd_h (64x64) x y (64x256)
#pragma unroll
        for (int nti = 0; nti < 4; ++nti) {
            int nt = wv + nti * 4;
#pragma unroll
            for (int mt = 0; mt < 4; ++mt) {
#pragma unroll
                for (int kt = 0; kt < 2; ++kt) {
                    FragBF a, b;
                    const u16* pa = Wdb + (((size_t)h * 64 + mt * 16 + l) * 64 + kt * 32 + hi * 8);
                    a.q[0] = *(const uint4*)pa;
                    a.q[1] = *(const uint4*)(pa + 16);
                    const u16* pb = &sYT[nt * 16 + l][kt * 32 + hi * 8];
                    b.q[0] = *(const uint4*)pb;
                    b.q[1] = *(const uint4*)(pb + 16);
                    acc[mt][nti] = __builtin_amdgcn_wmma_f32_16x16x32_bf16(
                        false, a.bf, false, b.bf, (short)0, acc[mt][nti], false, false);
                }
            }
        }
    }

    // store z (layout [n][o][t][25]); skip padded w >= 25
#pragma unroll
    for (int nti = 0; nti < 4; ++nti) {
        int nt = wv + nti * 4;
        int col = nt * 16 + l;
        int t = tc * CHT + (col >> 5);
        int w = col & 31;
        if (w < VV) {
#pragma unroll
            for (int mt = 0; mt < 4; ++mt) {
#pragma unroll
                for (int r = 0; r < 8; ++r) {
                    int o = mt * 16 + hi * 8 + r;
                    zout[(((size_t)n * OCN + o) * TT + t) * VV + w] = acc[mt][nti][r] + bdsum[o];
                }
            }
        }
    }
}

// ---------------------------------------------------------------------------
// K4: deterministic BN partial reduction: partial[o][slab][{sum,sumsq}]
// ---------------------------------------------------------------------------
__global__ __launch_bounds__(256) void k_bnstats(const float* __restrict__ z,
                                                 float* __restrict__ partial) {
    int o = blockIdx.x;
    int slab = blockIdx.y;  // 16 slabs of 2 n each
    float s = 0.f, ss = 0.f;
    for (int nn = slab * 2; nn < slab * 2 + 2; ++nn) {
        const float* p = z + ((size_t)nn * OCN + o) * (TT * VV);
        for (int i = threadIdx.x; i < TT * VV; i += 256) {
            float v = p[i];
            s += v;
            ss += v * v;
        }
    }
    __shared__ float rs[256], rss[256];
    rs[threadIdx.x] = s;
    rss[threadIdx.x] = ss;
    __syncthreads();
    for (int st = 128; st > 0; st >>= 1) {
        if (threadIdx.x < st) {
            rs[threadIdx.x] += rs[threadIdx.x + st];
            rss[threadIdx.x] += rss[threadIdx.x + st];
        }
        __syncthreads();
    }
    if (threadIdx.x == 0) {
        partial[(o * 16 + slab) * 2 + 0] = rs[0];
        partial[(o * 16 + slab) * 2 + 1] = rss[0];
    }
}

__global__ __launch_bounds__(64) void k_bnfinal(const float* __restrict__ partial,
                                                float* __restrict__ stats) {
    int o = threadIdx.x;
    float s = 0.f, ss = 0.f;
    for (int k = 0; k < 16; ++k) {
        s += partial[(o * 16 + k) * 2 + 0];
        ss += partial[(o * 16 + k) * 2 + 1];
    }
    stats[o] = s;
    stats[64 + o] = ss;
}

// ---------------------------------------------------------------------------
// K5: normalize + gamma/beta + residual + ReLU (in place on d_out)
// ---------------------------------------------------------------------------
__global__ __launch_bounds__(256) void k_bn_apply(float* __restrict__ zy,
                                                  const float* __restrict__ x,
                                                  const float* __restrict__ stats,
                                                  const float* __restrict__ gamma,
                                                  const float* __restrict__ beta) {
    size_t idx = (size_t)blockIdx.x * 256 + threadIdx.x;
    if (idx >= (size_t)NN * OCN * TT * VV) return;
    int o = (int)((idx / (TT * VV)) % OCN);
    float mean = stats[o] / BN_CNT;
    float var = stats[64 + o] / BN_CNT - mean * mean;
    float z = zy[idx];
    float y = gamma[o] * (z - mean) * rsqrtf(var + EPS) + beta[o];
    float r = y + x[idx];
    zy[idx] = r > 0.f ? r : 0.f;
}

// ---------------------------------------------------------------------------
extern "C" void kernel_launch(void* const* d_in, const int* in_sizes, int n_in,
                              void* d_out, int out_size, void* d_ws, size_t ws_size,
                              hipStream_t stream) {
    const float* x = (const float*)d_in[0];
    const float* Wa = (const float*)d_in[1];
    const float* ba = (const float*)d_in[2];
    const float* Wb = (const float*)d_in[3];
    const float* bb = (const float*)d_in[4];
    const float* Wd = (const float*)d_in[5];
    const float* bd = (const float*)d_in[6];
    const float* DepM = (const float*)d_in[7];
    const float* gamma = (const float*)d_in[8];
    const float* beta = (const float*)d_in[9];
    float* out = (float*)d_out;

    char* w = (char*)d_ws;
    size_t off = 0;
    auto alloc = [&](size_t bytes) -> void* {
        void* p = w + off;
        off = (off + bytes + 255) & ~(size_t)255;
        return p;
    };
    u16* xb_cm = (u16*)alloc((size_t)NN * TV * 64 * sizeof(u16));
    u16* xb_pad = (u16*)alloc((size_t)NN * CC * TT * 32 * sizeof(u16));
    u16* P = (u16*)alloc((size_t)HH * 32 * 64 * sizeof(u16));
    u16* Wdb = (u16*)alloc((size_t)HH * 64 * 64 * sizeof(u16));
    float* biasAB = (float*)alloc((size_t)HH * 32 * sizeof(float));
    float* bdsum = (float*)alloc(64 * sizeof(float));
    float* scorespad = (float*)alloc((size_t)NN * HH * 1024 * sizeof(float));
    u16* attnT = (u16*)alloc((size_t)NN * HH * 1024 * sizeof(u16));
    float* partial = (float*)alloc(64 * 16 * 2 * sizeof(float));
    float* stats = (float*)alloc(128 * sizeof(float));

    k_convert_x<<<dim3(NN * TT), 64, 0, stream>>>(x, xb_cm, xb_pad);
    k_convert_w<<<dim3(194), 256, 0, stream>>>(Wa, Wb, ba, bb, Wd, bd, P, Wdb, biasAB, bdsum);
    k_scores<<<dim3(NN * HH), 128, 0, stream>>>(xb_cm, P, biasAB, scorespad);
    k_softmax<<<dim3(NN * HH), 32, 0, stream>>>(scorespad, DepM, attnT);
    k_agg_out<<<dim3(NN, TT / CHT), 128, 0, stream>>>(xb_pad, attnT, Wdb, bdsum, out);
    k_bnstats<<<dim3(OCN, 16), 256, 0, stream>>>(out, partial);
    k_bnfinal<<<dim3(1), 64, 0, stream>>>(partial, stats);
    k_bn_apply<<<dim3((NN * OCN * TT * VV + 255) / 256), 256, 0, stream>>>(out, x, stats, gamma,
                                                                           beta);
}
